// SENDModel_23905787970075
// MI455X (gfx1250) — compile-verified, matrix operand order checked by reading, not run on hardware
//
#include <hip/hip_runtime.h>
#include <stdint.h>

typedef __attribute__((ext_vector_type(16))) _Float16 v16h;
typedef __attribute__((ext_vector_type(8)))  _Float16 v8h;
typedef __attribute__((ext_vector_type(4)))  _Float16 v4h;
typedef __attribute__((ext_vector_type(8)))  float    v8f;
typedef __attribute__((ext_vector_type(4)))  float    v4f;

#define TM 64
#define TN 64
#define TK 32
#define LDT 40   // padded LDS row stride in f16 (80B = 5*16B, keeps 16B alignment)

// ---------------------------------------------------------------------------
// Batched GEMM: C = act(A[f32, MxK] x Bt[f16, NxK]^T + bias + res)
// PRECONDITIONS (enforced by host-side padding):
//   - K is a multiple of 32 and operands are zero-padded in K
//   - Bt allocations are readable (zero-padded) up to ceil(N/64)*64 rows
//   - A allocations are readable up to ceil(M/64)*64 rows (values past M
//     are don't-care: epilogue guards stores with gm < M)
// No guarded staging -> branch-free inner loop, pure wide loads + WMMA.
// ---------------------------------------------------------------------------
__global__ __launch_bounds__(128) void gemm_wmma(
    const float* __restrict__ A, const _Float16* __restrict__ Bt,
    const float* __restrict__ bias, const float* __restrict__ res,
    float* __restrict__ C,
    int M, int N, int K, int lda, int ldb, int ldc, int ldres,
    int binner,
    int64_t sAi, int64_t sAo, int64_t sBi, int64_t sBo,
    int64_t sCi, int64_t sCo, int64_t sRi, int64_t sRo,
    int relu)
{
  __shared__ __align__(16) _Float16 As[TM * LDT];
  __shared__ __align__(16) _Float16 Bs[TN * LDT];

  const int bz = blockIdx.z;
  const int zo = bz / binner, zi = bz - zo * binner;
  A  += zo * sAo + zi * sAi;
  Bt += zo * sBo + zi * sBi;
  C  += zo * sCo + zi * sCi;
  if (res) res += zo * sRo + zi * sRi;

  const int tid  = threadIdx.x;
  const int wave = tid >> 5;
  const int lane = tid & 31;
  const int m0 = blockIdx.y * TM;
  const int n0 = blockIdx.x * TN;

  v8f acc0 = {}, acc1 = {}, acc2 = {}, acc3 = {};

  const int cbase = (lane & 16) ? 8 : 0;      // K sub-block per lane half
  const int mloc  = wave * 16 + (lane & 15);  // A row inside tile for this wave

  // staging coordinates
  const int ar = tid >> 3;            // 0..15   (A: 4 row-passes of 16)
  const int ac = (tid & 7) * 4;       // 0..28   (A: 4 f32 per thread)
  const int br = tid >> 2;            // 0..31   (B: 2 row-passes of 32)
  const int bc = (tid & 3) * 8;       // 0..24   (B: 8 f16 per thread)

  const float*    aptr = A  + (int64_t)(m0 + ar) * lda + ac;
  const _Float16* bptr = Bt + (int64_t)(n0 + br) * ldb + bc;

  for (int k0 = 0; k0 < K; k0 += TK) {
    // ---- stage A tile (64 x 32): 4x b128 f32 loads -> f16 -> ds_store_b64
    v4f a0 = *(const v4f*)(aptr);
    v4f a1 = *(const v4f*)(aptr + (int64_t)16 * lda);
    v4f a2 = *(const v4f*)(aptr + (int64_t)32 * lda);
    v4f a3 = *(const v4f*)(aptr + (int64_t)48 * lda);
    // ---- stage B tile (64 x 32): 2x 16B f16 loads
    v8h w0 = *(const v8h*)(bptr);
    v8h w1 = *(const v8h*)(bptr + (int64_t)32 * ldb);
    aptr += TK;
    bptr += TK;

    v4h h0 = { (_Float16)a0[0], (_Float16)a0[1], (_Float16)a0[2], (_Float16)a0[3] };
    v4h h1 = { (_Float16)a1[0], (_Float16)a1[1], (_Float16)a1[2], (_Float16)a1[3] };
    v4h h2 = { (_Float16)a2[0], (_Float16)a2[1], (_Float16)a2[2], (_Float16)a2[3] };
    v4h h3 = { (_Float16)a3[0], (_Float16)a3[1], (_Float16)a3[2], (_Float16)a3[3] };
    *(v4h*)&As[(ar     ) * LDT + ac] = h0;
    *(v4h*)&As[(ar + 16) * LDT + ac] = h1;
    *(v4h*)&As[(ar + 32) * LDT + ac] = h2;
    *(v4h*)&As[(ar + 48) * LDT + ac] = h3;
    *(v8h*)&Bs[(br     ) * LDT + bc] = w0;
    *(v8h*)&Bs[(br + 32) * LDT + bc] = w1;
    __syncthreads();

    // --- A fragment: lane holds row mloc, K = cbase+{0..7} and cbase+16+{0..7}
    const _Float16* ap = &As[mloc * LDT + cbase];
    v8h alo = *(const v8h*)(ap);
    v8h ahi = *(const v8h*)(ap + 16);
    v16h afrag = __builtin_shufflevector(alo, ahi, 0,1,2,3,4,5,6,7,8,9,10,11,12,13,14,15);

    // --- 4 B fragments / WMMAs: columns t*16 + (lane&15) ---
    #pragma unroll
    for (int t = 0; t < 4; t++) {
      const _Float16* bp = &Bs[(t * 16 + (lane & 15)) * LDT + cbase];
      v8h blo = *(const v8h*)(bp);
      v8h bhi = *(const v8h*)(bp + 16);
      v16h bfrag = __builtin_shufflevector(blo, bhi, 0,1,2,3,4,5,6,7,8,9,10,11,12,13,14,15);
      if      (t == 0) acc0 = __builtin_amdgcn_wmma_f32_16x16x32_f16(false, afrag, false, bfrag, (short)0, acc0, false, false);
      else if (t == 1) acc1 = __builtin_amdgcn_wmma_f32_16x16x32_f16(false, afrag, false, bfrag, (short)0, acc1, false, false);
      else if (t == 2) acc2 = __builtin_amdgcn_wmma_f32_16x16x32_f16(false, afrag, false, bfrag, (short)0, acc2, false, false);
      else             acc3 = __builtin_amdgcn_wmma_f32_16x16x32_f16(false, afrag, false, bfrag, (short)0, acc3, false, false);
    }
    __syncthreads();
  }

  // --- epilogue: D layout = VGPR r -> row r (+8 for lanes>=16), lane&15 -> col
  const int rowAdd = (lane & 16) ? 8 : 0;
  const int nloc   = lane & 15;
  #pragma unroll
  for (int t = 0; t < 4; t++) {
    int gn = n0 + t * 16 + nloc;
    if (gn >= N) continue;
    float bvv = bias ? bias[gn] : 0.0f;
    v8f a = (t == 0) ? acc0 : (t == 1) ? acc1 : (t == 2) ? acc2 : acc3;
    #pragma unroll
    for (int r = 0; r < 8; r++) {
      int gm = m0 + wave * 16 + r + rowAdd;
      if (gm < M) {
        float v = a[r] + bvv;
        if (res)  v += res[(int64_t)gm * ldres + gn];
        if (relu) v = v > 0.0f ? v : 0.0f;
        C[(int64_t)gm * ldc + gn] = v;
      }
    }
  }
}

// ---------------------------------------------------------------------------
// FSMN windowed mean with padded output stride:
// mean[b,t,d<Dn] = avg(x[b, max(0,t-stride)..t, d]); mean[b,t,Dn<=d<Dpad] = 0
// ---------------------------------------------------------------------------
__global__ void fsmn_mean_kernel(const float* __restrict__ x, float* __restrict__ mean,
                                 int Bn, int Tn, int Dn, int Dpad, int stride)
{
  int64_t i = (int64_t)blockIdx.x * blockDim.x + threadIdx.x;
  int64_t total = (int64_t)Bn * Tn * Dpad;
  if (i >= total) return;
  int d = (int)(i % Dpad);
  int64_t bt = i / Dpad;
  if (d >= Dn) { mean[i] = 0.0f; return; }
  int t = (int)(bt % Tn);
  int b = (int)(bt / Tn);
  int t0 = t - stride; if (t0 < 0) t0 = 0;
  const float* px = x + ((int64_t)b * Tn + t0) * Dn + d;
  float s = 0.0f;
  for (int tt = t0; tt <= t; tt++) { s += *px; px += Dn; }
  mean[i] = s / (float)(t - t0 + 1);
}

// ---------------------------------------------------------------------------
// LayerNorm over last dim (one block per row)
// ---------------------------------------------------------------------------
__global__ __launch_bounds__(128) void ln_kernel(const float* __restrict__ x,
    const float* __restrict__ g, const float* __restrict__ b,
    float* __restrict__ y, int Hd)
{
  __shared__ float red[128];
  int64_t row = blockIdx.x;
  const float* px = x + row * Hd;
  int tid = threadIdx.x;
  float s = 0.0f;
  for (int i = tid; i < Hd; i += 128) s += px[i];
  red[tid] = s; __syncthreads();
  for (int o = 64; o > 0; o >>= 1) { if (tid < o) red[tid] += red[tid + o]; __syncthreads(); }
  float m = red[0] * (1.0f / Hd);
  __syncthreads();
  float v = 0.0f;
  for (int i = tid; i < Hd; i += 128) { float d = px[i] - m; v += d * d; }
  red[tid] = v; __syncthreads();
  for (int o = 64; o > 0; o >>= 1) { if (tid < o) red[tid] += red[tid + o]; __syncthreads(); }
  float rs = rsqrtf(red[0] * (1.0f / Hd) + 1e-5f);
  float* py = y + row * Hd;
  for (int i = tid; i < Hd; i += 128) py[i] = (px[i] - m) * rs * g[i] + b[i];
}

// ---------------------------------------------------------------------------
// In-place scaled softmax over rows of length L (one block per row)
// ---------------------------------------------------------------------------
__global__ __launch_bounds__(256) void softmax_kernel(float* __restrict__ x, int L, float scale)
{
  __shared__ float red[256];
  float* px = x + (int64_t)blockIdx.x * L;
  int tid = threadIdx.x;
  float mx = -3.402823e38f;
  for (int i = tid; i < L; i += 256) mx = fmaxf(mx, px[i]);
  red[tid] = mx; __syncthreads();
  for (int o = 128; o > 0; o >>= 1) { if (tid < o) red[tid] = fmaxf(red[tid], red[tid + o]); __syncthreads(); }
  float m = red[0];
  __syncthreads();
  float s = 0.0f;
  for (int i = tid; i < L; i += 256) { float e = __expf((px[i] - m) * scale); px[i] = e; s += e; }
  red[tid] = s; __syncthreads();
  for (int o = 128; o > 0; o >>= 1) { if (tid < o) red[tid] += red[tid + o]; __syncthreads(); }
  float inv = 1.0f / red[0];
  for (int i = tid; i < L; i += 256) px[i] *= inv;
}

// ---------------------------------------------------------------------------
// Converters / padders
// ---------------------------------------------------------------------------
__global__ void cvt_f16_kernel(const float* __restrict__ s, _Float16* __restrict__ d, int64_t n)
{
  int64_t i = (int64_t)blockIdx.x * blockDim.x + threadIdx.x;
  if (i < n) d[i] = (_Float16)s[i];
}

// src[R][C] f32 -> dst[Cpad][ldd] f16, transposed with zero padding:
// dst[c][r] = (c<C && r<R) ? src[r][c] : 0
__global__ void cvt_f16_T_kernel(const float* __restrict__ s, _Float16* __restrict__ d,
                                 int R, int Cc, int ldd, int Cpad)
{
  int64_t i = (int64_t)blockIdx.x * blockDim.x + threadIdx.x;
  if (i >= (int64_t)Cpad * ldd) return;
  int r = (int)(i % ldd);
  int c = (int)(i / ldd);
  d[i] = (c < Cc && r < R) ? (_Float16)s[(int64_t)r * Cc + c] : (_Float16)0.0f;
}

// src[Rs][Cs] f32 -> dst[Rd][Cd] f32 zero-padded copy
__global__ void pad_f32_kernel(const float* __restrict__ s, float* __restrict__ d,
                               int Rs, int Cs, int Rd, int Cd)
{
  int64_t i = (int64_t)blockIdx.x * blockDim.x + threadIdx.x;
  if (i >= (int64_t)Rd * Cd) return;
  int c = (int)(i % Cd);
  int r = (int)(i / Cd);
  d[i] = (r < Rs && c < Cs) ? s[(int64_t)r * Cs + c] : 0.0f;
}

// s1[(b*Sn+r)][Hn] f32 -> dst[b][64][Hn] f16 (rows >= Sn zeroed)
__global__ void spkpad16_kernel(const float* __restrict__ s, _Float16* __restrict__ d,
                                int Bn, int Sn, int Hn)
{
  int64_t i = (int64_t)blockIdx.x * blockDim.x + threadIdx.x;
  if (i >= (int64_t)Bn * 64 * Hn) return;
  int c = (int)(i % Hn);
  int64_t rr = i / Hn;
  int r = (int)(rr % 64);
  int b = (int)(rr / 64);
  d[i] = (r < Sn) ? (_Float16)s[((int64_t)b * Sn + r) * Hn + c] : (_Float16)0.0f;
}

// v[B][T][NH*HD] f32 -> vT[(b*NH+h)][HD][T] f16
__global__ void vtrans_kernel(const float* __restrict__ v, _Float16* __restrict__ vT,
                              int Bn, int Tn, int NHn, int HDn)
{
  int64_t i = (int64_t)blockIdx.x * blockDim.x + threadIdx.x;
  int64_t total = (int64_t)Bn * NHn * HDn * Tn;
  if (i >= total) return;
  int t = (int)(i % Tn);
  int64_t r = i / Tn;
  int d = (int)(r % HDn); r /= HDn;
  int h = (int)(r % NHn);
  int b = (int)(r / NHn);
  vT[i] = (_Float16)v[((int64_t)b * Tn + t) * (NHn * HDn) + h * HDn + d];
}

// cc[row][Wpad] = concat(ci[row][S], h[row][H], zeros)
__global__ void concat_kernel(const float* __restrict__ ci, const float* __restrict__ h,
                              float* __restrict__ cc, int rows, int Sn, int Hn, int Wpad)
{
  int64_t i = (int64_t)blockIdx.x * blockDim.x + threadIdx.x;
  if (i >= (int64_t)rows * Wpad) return;
  int c = (int)(i % Wpad);
  int64_t r = i / Wpad;
  float v;
  if (c < Sn)           v = ci[r * Sn + c];
  else if (c < Sn + Hn) v = h[r * Hn + (c - Sn)];
  else                  v = 0.0f;
  cc[i] = v;
}

// ---------------------------------------------------------------------------
// Host orchestration
// ---------------------------------------------------------------------------
static void launch_gemm(hipStream_t st, const float* A, const _Float16* Bt, const float* bias,
                        const float* res, float* C, int M, int N, int K,
                        int lda, int ldb, int ldc, int ldres, int relu,
                        int batch = 1, int binner = 1,
                        int64_t sAi = 0, int64_t sAo = 0, int64_t sBi = 0, int64_t sBo = 0,
                        int64_t sCi = 0, int64_t sCo = 0, int64_t sRi = 0, int64_t sRo = 0)
{
  dim3 grid((N + TN - 1) / TN, (M + TM - 1) / TM, batch);
  gemm_wmma<<<grid, 128, 0, st>>>(A, Bt, bias, res, C, M, N, K, lda, ldb, ldc, ldres,
                                  binner, sAi, sAo, sBi, sBo, sCi, sCo, sRi, sRo, relu);
}

static void launch_cvtT(hipStream_t st, const float* s, _Float16* d, int R, int Cc, int ldd, int Cpad)
{
  int64_t n = (int64_t)Cpad * ldd;
  cvt_f16_T_kernel<<<(unsigned)((n + 255) / 256), 256, 0, st>>>(s, d, R, Cc, ldd, Cpad);
}

static void launch_cvt(hipStream_t st, const float* s, _Float16* d, int64_t n)
{
  cvt_f16_kernel<<<(unsigned)((n + 255) / 256), 256, 0, st>>>(s, d, n);
}

extern "C" void kernel_launch(void* const* d_in, const int* in_sizes, int n_in,
                              void* d_out, int out_size, void* d_ws, size_t ws_size,
                              hipStream_t stream)
{
  const int Bn = 4, Tn = 1024, Sn = 4, DIN = 80, Hn = 512, SPKD = 192, FFn = 2048;
  const int NHn = 4, HDn = 128, NLn = 4;
  const int MT = Bn * Tn;              // 4096 rows
  const int HH = Hn * Hn;
  const int DINP = 96;                 // DIN padded to multiple of 32
  const int KCMB = Sn + Hn;            // 516
  const int KCMBP = 544;               // padded to multiple of 32

  const float* x      = (const float*)d_in[0];
  const float* spkemb = (const float*)d_in[1];
  const float* e0W1 = (const float*)d_in[2];  const float* e0b1 = (const float*)d_in[3];
  const float* e0Wm = (const float*)d_in[4];  const float* e0bm = (const float*)d_in[5];
  const float* eW1  = (const float*)d_in[6];  const float* eb1  = (const float*)d_in[7];
  const float* eWm  = (const float*)d_in[8];  const float* ebm  = (const float*)d_in[9];
  const float* sW0 = (const float*)d_in[10];  const float* sb0 = (const float*)d_in[11];
  const float* sW1 = (const float*)d_in[12];  const float* sb1 = (const float*)d_in[13];
  const float* sW2 = (const float*)d_in[14];  const float* sb2 = (const float*)d_in[15];
  const float* Wq = (const float*)d_in[16];   const float* bq = (const float*)d_in[17];
  const float* Wk = (const float*)d_in[18];   const float* bk = (const float*)d_in[19];
  const float* Wv = (const float*)d_in[20];   const float* bv = (const float*)d_in[21];
  const float* Wo = (const float*)d_in[22];   const float* bo = (const float*)d_in[23];
  const float* ln1g = (const float*)d_in[24]; const float* ln1b = (const float*)d_in[25];
  const float* W1 = (const float*)d_in[26];   const float* b1 = (const float*)d_in[27];
  const float* W2 = (const float*)d_in[28];   const float* b2 = (const float*)d_in[29];
  const float* ln2g = (const float*)d_in[30]; const float* ln2b = (const float*)d_in[31];
  const float* combW = (const float*)d_in[32]; const float* combB = (const float*)d_in[33];
  const float* pW1 = (const float*)d_in[34];  const float* pb1 = (const float*)d_in[35];
  const float* pWm = (const float*)d_in[36];  const float* pbm = (const float*)d_in[37];
  const float* cW1 = (const float*)d_in[38];  const float* cb1 = (const float*)d_in[39];
  const float* cW2 = (const float*)d_in[40];  const float* cb2 = (const float*)d_in[41];
  (void)in_sizes; (void)n_in; (void)out_size; (void)ws_size;

  // bump allocator over workspace
  char* base = (char*)d_ws;
  size_t off = 0;
  auto alloc = [&](size_t bytes) -> char* {
    char* p = base + off;
    off += bytes; off = (off + 255) & ~(size_t)255;
    return p;
  };

  // ----- f16 transposed weights (zero-padded in K / N-rows) -----
  _Float16* wE0W1 = (_Float16*)alloc(sizeof(_Float16) * (size_t)Hn * DINP);
  _Float16* wE0Wm = (_Float16*)alloc(sizeof(_Float16) * (size_t)Hn * DINP);
  _Float16* wEW1  = (_Float16*)alloc(sizeof(_Float16) * 7 * (size_t)HH);
  _Float16* wEWm  = (_Float16*)alloc(sizeof(_Float16) * 7 * (size_t)HH);
  _Float16* wS0   = (_Float16*)alloc(sizeof(_Float16) * (size_t)Hn * SPKD);
  _Float16* wS1   = (_Float16*)alloc(sizeof(_Float16) * (size_t)HH);
  _Float16* wS2   = (_Float16*)alloc(sizeof(_Float16) * (size_t)HH);
  _Float16* wQ    = (_Float16*)alloc(sizeof(_Float16) * NLn * (size_t)HH);
  _Float16* wK    = (_Float16*)alloc(sizeof(_Float16) * NLn * (size_t)HH);
  _Float16* wV    = (_Float16*)alloc(sizeof(_Float16) * NLn * (size_t)HH);
  _Float16* wO    = (_Float16*)alloc(sizeof(_Float16) * NLn * (size_t)HH);
  _Float16* wF1   = (_Float16*)alloc(sizeof(_Float16) * NLn * (size_t)Hn * FFn);
  _Float16* wF2   = (_Float16*)alloc(sizeof(_Float16) * NLn * (size_t)Hn * FFn);
  _Float16* wCmb  = (_Float16*)alloc(sizeof(_Float16) * (size_t)Hn * KCMBP);
  _Float16* wPW1  = (_Float16*)alloc(sizeof(_Float16) * 6 * (size_t)HH);
  _Float16* wPWm  = (_Float16*)alloc(sizeof(_Float16) * 6 * (size_t)HH);
  _Float16* wC1   = (_Float16*)alloc(sizeof(_Float16) * (size_t)HH);
  _Float16* wC2   = (_Float16*)alloc(sizeof(_Float16) * 64 * (size_t)Hn);   // N=16 padded to 64 rows

  // ----- f16 activation buffers -----
  _Float16* spk16 = (_Float16*)alloc(sizeof(_Float16) * (size_t)Bn * 64 * Hn);  // N=4 padded to 64 rows/batch
  _Float16* k16   = (_Float16*)alloc(sizeof(_Float16) * (size_t)MT * Hn);
  _Float16* vT16  = (_Float16*)alloc(sizeof(_Float16) * (size_t)Bn * NHn * HDn * Tn);

  // ----- f32 activation buffers -----
  float* xpad = (float*)alloc(sizeof(float) * (size_t)MT * DINP);
  float* spkp = (float*)alloc(sizeof(float) * (size_t)64 * SPKD);    // M=16 padded to 64 rows
  float* actA = (float*)alloc(sizeof(float) * (size_t)MT * Hn);
  float* actB = (float*)alloc(sizeof(float) * (size_t)MT * Hn);
  float* qb   = (float*)alloc(sizeof(float) * (size_t)MT * Hn);      // also FSMN mean buf
  float* kb   = (float*)alloc(sizeof(float) * (size_t)MT * Hn);
  float* vb   = (float*)alloc(sizeof(float) * (size_t)MT * Hn);
  float* ob   = (float*)alloc(sizeof(float) * (size_t)MT * Hn);
  float* t1   = (float*)alloc(sizeof(float) * (size_t)MT * Hn);
  float* ffb  = (float*)alloc(sizeof(float) * (size_t)MT * FFn);     // also concat buf
  float* ci   = (float*)alloc(sizeof(float) * (size_t)MT * Sn);
  float* s1   = (float*)alloc(sizeof(float) * (size_t)64 * Hn);
  float* s2   = (float*)alloc(sizeof(float) * (size_t)64 * Hn);
  float* scor = (float*)alloc(sizeof(float) * (size_t)Bn * NHn * Tn * Tn);
  float* meanb = qb;
  float* ccb   = ffb;

  // ===== weight conversion (transpose to [N][Kpad] f16, zero-padded) =====
  launch_cvtT(stream, e0W1, wE0W1, DIN, Hn, DINP, Hn);
  launch_cvtT(stream, e0Wm, wE0Wm, DIN, Hn, DINP, Hn);
  for (int i = 0; i < 7; i++) {
    launch_cvtT(stream, eW1 + (size_t)i * HH, wEW1 + (size_t)i * HH, Hn, Hn, Hn, Hn);
    launch_cvtT(stream, eWm + (size_t)i * HH, wEWm + (size_t)i * HH, Hn, Hn, Hn, Hn);
  }
  launch_cvtT(stream, sW0, wS0, SPKD, Hn, SPKD, Hn);
  launch_cvtT(stream, sW1, wS1, Hn, Hn, Hn, Hn);
  launch_cvtT(stream, sW2, wS2, Hn, Hn, Hn, Hn);
  for (int l = 0; l < NLn; l++) {
    launch_cvtT(stream, Wq + (size_t)l * HH, wQ + (size_t)l * HH, Hn, Hn, Hn, Hn);
    launch_cvtT(stream, Wk + (size_t)l * HH, wK + (size_t)l * HH, Hn, Hn, Hn, Hn);
    launch_cvtT(stream, Wv + (size_t)l * HH, wV + (size_t)l * HH, Hn, Hn, Hn, Hn);
    launch_cvtT(stream, Wo + (size_t)l * HH, wO + (size_t)l * HH, Hn, Hn, Hn, Hn);
    launch_cvtT(stream, W1 + (size_t)l * Hn * FFn, wF1 + (size_t)l * Hn * FFn, Hn, FFn, Hn, FFn);
    launch_cvtT(stream, W2 + (size_t)l * Hn * FFn, wF2 + (size_t)l * Hn * FFn, FFn, Hn, FFn, Hn);
  }
  launch_cvtT(stream, combW, wCmb, KCMB, Hn, KCMBP, Hn);
  for (int i = 0; i < 6; i++) {
    launch_cvtT(stream, pW1 + (size_t)i * HH, wPW1 + (size_t)i * HH, Hn, Hn, Hn, Hn);
    launch_cvtT(stream, pWm + (size_t)i * HH, wPWm + (size_t)i * HH, Hn, Hn, Hn, Hn);
  }
  launch_cvtT(stream, cW1, wC1, Hn, Hn, Hn, Hn);
  launch_cvtT(stream, cW2, wC2, Hn, 16, Hn, 64);

  // ===== input padding =====
  {
    int64_t n = (int64_t)MT * DINP;
    pad_f32_kernel<<<(unsigned)((n + 255) / 256), 256, 0, stream>>>(x, xpad, MT, DIN, MT, DINP);
    int64_t n2 = (int64_t)64 * SPKD;
    pad_f32_kernel<<<(unsigned)((n2 + 255) / 256), 256, 0, stream>>>(spkemb, spkp, Bn * Sn, SPKD, 64, SPKD);
  }

  // ===== speech encoder: 8 FSMN layers =====
  {
    int64_t n0 = (int64_t)MT * DINP;
    fsmn_mean_kernel<<<(unsigned)((n0 + 255) / 256), 256, 0, stream>>>(x, meanb, Bn, Tn, DIN, DINP, 1);
  }
  launch_gemm(stream, xpad,  wE0W1, e0b1, nullptr, actA, MT, Hn, DINP, DINP, DINP, Hn, Hn, 0);
  launch_gemm(stream, meanb, wE0Wm, e0bm, actA,    actA, MT, Hn, DINP, DINP, DINP, Hn, Hn, 0);

  float* cur = actA; float* nxt = actB;
  int64_t nH = (int64_t)MT * Hn;
  for (int i = 0; i < 7; i++) {
    int stride = 2 << i;   // 2^(i+1)
    fsmn_mean_kernel<<<(unsigned)((nH + 255) / 256), 256, 0, stream>>>(cur, meanb, Bn, Tn, Hn, Hn, stride);
    launch_gemm(stream, cur,   wEW1 + (size_t)i * HH, eb1 + (size_t)i * Hn, nullptr, nxt, MT, Hn, Hn, Hn, Hn, Hn, Hn, 0);
    launch_gemm(stream, meanb, wEWm + (size_t)i * HH, ebm + (size_t)i * Hn, nxt,    nxt, MT, Hn, Hn, Hn, Hn, Hn, Hn, 0);
    float* tmp = cur; cur = nxt; nxt = tmp;
  }

  // ===== speaker MLP + CI scores =====
  launch_gemm(stream, spkp, wS0, sb0, nullptr, s1, Bn * Sn, Hn, SPKD, SPKD, SPKD, Hn, Hn, 1);
  launch_gemm(stream, s1,   wS1, sb1, nullptr, s2, Bn * Sn, Hn, Hn,   Hn,   Hn,   Hn, Hn, 1);
  launch_gemm(stream, s2,   wS2, sb2, nullptr, s1, Bn * Sn, Hn, Hn,   Hn,   Hn,   Hn, Hn, 0);
  {
    int64_t n = (int64_t)Bn * 64 * Hn;
    spkpad16_kernel<<<(unsigned)((n + 255) / 256), 256, 0, stream>>>(s1, spk16, Bn, Sn, Hn);
  }
  launch_gemm(stream, cur, spk16, nullptr, nullptr, ci, Tn, Sn, Hn, Hn, Hn, Sn, Sn, 0,
              /*batch=*/Bn, /*binner=*/1,
              0, (int64_t)Tn * Hn, 0, (int64_t)64 * Hn, 0, (int64_t)Tn * Sn, 0, 0);

  // ===== transformer (post-LN), h lives in `cur` =====
  float* h = cur;
  const float scale = 1.0f / 11.3137084989847604f;  // 1/sqrt(128)
  for (int l = 0; l < NLn; l++) {
    launch_gemm(stream, h, wQ + (size_t)l * HH, bq + (size_t)l * Hn, nullptr, qb, MT, Hn, Hn, Hn, Hn, Hn, Hn, 0);
    launch_gemm(stream, h, wK + (size_t)l * HH, bk + (size_t)l * Hn, nullptr, kb, MT, Hn, Hn, Hn, Hn, Hn, Hn, 0);
    launch_gemm(stream, h, wV + (size_t)l * HH, bv + (size_t)l * Hn, nullptr, vb, MT, Hn, Hn, Hn, Hn, Hn, Hn, 0);
    launch_cvt(stream, kb, k16, nH);
    {
      int64_t nv = (int64_t)Bn * NHn * HDn * Tn;
      vtrans_kernel<<<(unsigned)((nv + 255) / 256), 256, 0, stream>>>(vb, vT16, Bn, Tn, NHn, HDn);
    }
    // scores[z] = Q_z @ K_z^T   (z = b*NH + head)
    launch_gemm(stream, qb, k16, nullptr, nullptr, scor, Tn, Tn, HDn, Hn, Hn, Tn, Tn, 0,
                /*batch=*/Bn * NHn, /*binner=*/NHn,
                /*sAi=*/HDn, /*sAo=*/(int64_t)Tn * Hn,
                /*sBi=*/HDn, /*sBo=*/(int64_t)Tn * Hn,
                /*sCi=*/(int64_t)Tn * Tn, /*sCo=*/(int64_t)NHn * Tn * Tn, 0, 0);
    softmax_kernel<<<Bn * NHn * Tn, 256, 0, stream>>>(scor, Tn, scale);
    // O[z] = P_z @ V_z
    launch_gemm(stream, scor, vT16, nullptr, nullptr, ob, Tn, HDn, Tn, Tn, Tn, Hn, Hn, 0,
                /*batch=*/Bn * NHn, /*binner=*/NHn,
                /*sAi=*/(int64_t)Tn * Tn, /*sAo=*/(int64_t)NHn * Tn * Tn,
                /*sBi=*/(int64_t)HDn * Tn, /*sBo=*/(int64_t)NHn * HDn * Tn,
                /*sCi=*/HDn, /*sCo=*/(int64_t)Tn * Hn, 0, 0);
    // h = LN(h + O @ Wo + bo)
    launch_gemm(stream, ob, wO + (size_t)l * HH, bo + (size_t)l * Hn, h, t1, MT, Hn, Hn, Hn, Hn, Hn, Hn, 0);
    ln_kernel<<<MT, 128, 0, stream>>>(t1, ln1g + (size_t)l * Hn, ln1b + (size_t)l * Hn, h, Hn);
    // h = LN(h + relu(h@W1+b1)@W2 + b2)
    launch_gemm(stream, h,   wF1 + (size_t)l * Hn * FFn, b1 + (size_t)l * FFn, nullptr, ffb, MT, FFn, Hn, Hn, Hn, FFn, FFn, 1);
    launch_gemm(stream, ffb, wF2 + (size_t)l * Hn * FFn, b2 + (size_t)l * Hn,  h,       t1,  MT, Hn, FFn, FFn, FFn, Hn, Hn, 0);
    ln_kernel<<<MT, 128, 0, stream>>>(t1, ln2g + (size_t)l * Hn, ln2b + (size_t)l * Hn, h, Hn);
  }

  // ===== combine + 6 post-FSMN layers + classifier =====
  {
    int64_t nc = (int64_t)MT * KCMBP;
    concat_kernel<<<(unsigned)((nc + 255) / 256), 256, 0, stream>>>(ci, h, ccb, MT, Sn, Hn, KCMBP);
  }
  float* pcur = (h == actA) ? actB : actA;
  float* pnxt = h;
  launch_gemm(stream, ccb, wCmb, combB, nullptr, pcur, MT, Hn, KCMBP, KCMBP, KCMBP, Hn, Hn, 0);
  for (int i = 0; i < 6; i++) {
    int stride = 1 << i;   // 2^i
    fsmn_mean_kernel<<<(unsigned)((nH + 255) / 256), 256, 0, stream>>>(pcur, meanb, Bn, Tn, Hn, Hn, stride);
    launch_gemm(stream, pcur,  wPW1 + (size_t)i * HH, pb1 + (size_t)i * Hn, nullptr, pnxt, MT, Hn, Hn, Hn, Hn, Hn, Hn, 0);
    launch_gemm(stream, meanb, wPWm + (size_t)i * HH, pbm + (size_t)i * Hn, pnxt,    pnxt, MT, Hn, Hn, Hn, Hn, Hn, Hn, 0);
    float* tmp = pcur; pcur = pnxt; pnxt = tmp;
  }
  launch_gemm(stream, pcur, wC1, cb1, nullptr, t1, MT, Hn, Hn, Hn, Hn, Hn, Hn, 1);
  launch_gemm(stream, t1,   wC2, cb2, nullptr, (float*)d_out, MT, 16, Hn, Hn, Hn, 16, 16, 0);
}
